// MDU_15083925144367
// MI455X (gfx1250) — matrix operand order repaired and checked by reading, not compiled, not run on hardware
//
#include <hip/hip_runtime.h>

// ---------------- types ----------------
typedef __bf16 bf16;
typedef __attribute__((ext_vector_type(16))) __bf16    v16bf;
typedef __attribute__((ext_vector_type(8)))  float     v8f;
typedef __attribute__((ext_vector_type(4)))  unsigned  u32x4;

union Frag { v16bf v; u32x4 q[2]; };

__device__ __forceinline__ v8f zero8() {
    v8f z;
#pragma unroll
    for (int i = 0; i < 8; ++i) z[i] = 0.f;
    return z;
}

__device__ __forceinline__ v8f wmma_bf16(const Frag& a, const Frag& b, v8f c) {
    // D = A(16x32 bf16) * B(32x16 bf16) + C(16x16 f32)
    return __builtin_amdgcn_wmma_f32_16x16x32_bf16(
        false, a.v, false, b.v, (short)0, c, false, false);
}

// ---------------- problem constants ----------------
#define CN   64            // channels (xC = yC = outC)
#define HH   128
#define WW   128
#define SPX  72            // LDS bf16 stride per pixel (64 + 8 pad -> 36 dwords, bank-spread)
#define ROWPX 130          // W + 2 halo

// workspace byte offsets
#define WS_W1P 0u
#define WS_W3P 73728u
#define WS_W2P 147456u
#define WS_B2P 221184u
#define WS_R2  223488u

// =====================================================================
// prep: repack weights into WMMA B-fragment register order.
// B frag (16x16x32 bf16): lane l holds column n = n0 + (l&15),
// K values kc*32 + ((l>=16)?8:0) + {[0..8), [16..24)}, dword v = (K=2v, K=2v+1) of that list.
// 1KB block per (tap, kchunk, ntile); lane's 32B contiguous at lane*32.
// =====================================================================
__global__ __launch_bounds__(256) void MDU_prep_pack(
    const float* __restrict__ w1, const float* __restrict__ w2,
    const float* __restrict__ w3, const float* __restrict__ b2,
    unsigned* __restrict__ w1p, unsigned* __restrict__ w2p,
    unsigned* __restrict__ w3p, float* __restrict__ b2p)
{
    const int b = blockIdx.x, tid = threadIdx.x;
    const int lane = tid >> 3, v = tid & 7;
    if (b < 144) {                       // w1 (b<72) and w3 (b>=72): OIHW 3x3 convs
        const float* w = (b < 72) ? w1 : w3;
        unsigned* dst  = (b < 72) ? w1p : w3p;
        const int bb = b % 72;
        const int t = bb >> 3, kc = (bb >> 2) & 1, nt = bb & 3;
        const int k = kc * 32 + ((lane >= 16) ? 8 : 0) + ((v >= 4) ? 16 : 0) + 2 * (v & 3);
        const int n = nt * 16 + (lane & 15);
        union { __bf16 h[2]; unsigned u; } p;
        p.h[0] = (__bf16)w[(n * CN + k) * 9 + t];
        p.h[1] = (__bf16)w[(n * CN + k + 1) * 9 + t];
        dst[bb * 256 + tid] = p.u;
    } else if (b < 216) {                // w2: [576][64] 1x1, N permuted to n' = t*64 + c
        const int bb = b - 144;
        const int ntp = bb >> 1, kc = bb & 1;
        const int k  = kc * 32 + ((lane >= 16) ? 8 : 0) + ((v >= 4) ? 16 : 0) + 2 * (v & 3);
        const int np = ntp * 16 + (lane & 15);
        const int t = np >> 6, c = np & 63;
        union { __bf16 h[2]; unsigned u; } p;
        p.h[0] = (__bf16)w2[(c * 9 + t) * CN + k];
        p.h[1] = (__bf16)w2[(c * 9 + t) * CN + k + 1];
        w2p[bb * 256 + tid] = p.u;
    } else {                             // b2 permuted to [t][c]
        for (int i = tid; i < 576; i += 256) {
            const int t = i >> 6, c = i & 63;
            b2p[i] = b2[c * 9 + t];
        }
    }
}

// =====================================================================
// k12: fused conv3x3(y)+b1 -> 1x1 GEMM (64->576, tap-major N) + bias
//      -> dynamic depthwise with x -> res (channels-last bf16).
// One block per (n,h) output row; 8 waves x 16 pixels.
// =====================================================================
__global__ __launch_bounds__(256, 2) void MDU_k12_fused(
    const float* __restrict__ x, const float* __restrict__ y,
    const unsigned* __restrict__ w1p, const float* __restrict__ b1,
    const unsigned* __restrict__ w2p, const float* __restrict__ b2p,
    bf16* __restrict__ r2)
{
    extern __shared__ char smem[];
    bf16* ldsY = (bf16*)smem;                               // [3*130][SPX]
    bf16* ldsX = (bf16*)(smem + 3 * ROWPX * SPX * 2);       // [3*130][SPX]
    bf16* ldsK = (bf16*)(smem + 2 * 3 * ROWPX * SPX * 2);   // [128][SPX]

    const int n = blockIdx.x >> 7, h = blockIdx.x & 127;
    const int tid = threadIdx.x;

    // ---- stage y and x rows h-1..h+1 (NCHW f32 -> [dh][px][c] bf16) ----
    for (int idx = tid; idx < 3 * CN * ROWPX; idx += 256) {
        const int px = idx % ROWPX;
        const int r  = idx / ROWPX;
        const int c  = r % CN, dh = r / CN;
        const int gh = h + dh - 1, gw = px - 1;
        const bool ok = ((unsigned)gh < 128u) && ((unsigned)gw < 128u);
        const int base = (((n * CN + c) * HH + gh) * WW + gw);
        const float vy = ok ? y[base] : 0.f;
        const float vx = ok ? x[base] : 0.f;
        ldsY[(dh * ROWPX + px) * SPX + c] = (bf16)vy;
        ldsX[(dh * ROWPX + px) * SPX + c] = (bf16)vx;
    }
    __syncthreads();

    const int lane = tid & 31;
    const int pb   = (tid >> 5) << 4;       // wave pixel base
    const int mA   = lane & 15;             // A-frag row
    const int klb  = (lane >> 4) << 3;      // A-frag K sub-base (0/8)
    const int colo = lane & 15;             // C/D column
    const int mhi  = (lane >> 4) << 3;      // C/D row offset (0/8)

    // ---- conv1: 9 taps x 2 k-chunks x 4 n-tiles of WMMA ----
    v8f acc[4];
#pragma unroll
    for (int i = 0; i < 4; ++i) acc[i] = zero8();

    for (int t = 0; t < 9; ++t) {
        const int dh = t / 3, dw = t % 3;
#pragma unroll
        for (int kc = 0; kc < 2; ++kc) {
            const bf16* ap = ldsY + (dh * ROWPX + pb + mA + dw) * SPX + kc * 32 + klb;
            Frag A;
            A.q[0] = *(const u32x4*)(ap);
            A.q[1] = *(const u32x4*)(ap + 16);
#pragma unroll
            for (int nt = 0; nt < 4; ++nt) {
                const unsigned* bp = w1p + (((t * 2 + kc) * 4 + nt) << 8) + (lane << 3);
                Frag B;
                B.q[0] = *(const u32x4*)(bp);
                B.q[1] = *(const u32x4*)(bp + 4);
                acc[nt] = wmma_bf16(A, B, acc[nt]);
            }
        }
    }

    // ---- k1 row (+bias) -> LDS, channels-minor bf16 ----
#pragma unroll
    for (int nt = 0; nt < 4; ++nt) {
        const float bv = b1[nt * 16 + colo];
#pragma unroll
        for (int r = 0; r < 8; ++r) {
            const int px = pb + r + mhi;
            ldsK[px * SPX + nt * 16 + colo] = (bf16)(acc[nt][r] + bv);
        }
    }
    __syncthreads();

    // ---- conv2 (1x1) + dynamic depthwise ----
    Frag A0, A1;   // two K-chunks of this wave's 16 pixels (invariant over t,nt)
    {
        const bf16* ap = ldsK + (pb + mA) * SPX + klb;
        A0.q[0] = *(const u32x4*)(ap);        A0.q[1] = *(const u32x4*)(ap + 16);
        A1.q[0] = *(const u32x4*)(ap + 32);   A1.q[1] = *(const u32x4*)(ap + 48);
    }

    v8f res[4];
#pragma unroll
    for (int i = 0; i < 4; ++i) res[i] = zero8();

    for (int t = 0; t < 9; ++t) {
        const int dh = t / 3, dw = t % 3;
#pragma unroll
        for (int nt = 0; nt < 4; ++nt) {
            const int ntp = t * 4 + nt;          // permuted N tile: tap-major
            v8f kv = zero8();
            const unsigned* bp = w2p + ((ntp * 2) << 8) + (lane << 3);
            Frag B;
            B.q[0] = *(const u32x4*)(bp);
            B.q[1] = *(const u32x4*)(bp + 4);
            kv = wmma_bf16(A0, B, kv);
            bp += 256;
            B.q[0] = *(const u32x4*)(bp);
            B.q[1] = *(const u32x4*)(bp + 4);
            kv = wmma_bf16(A1, B, kv);

            const float bv = b2p[t * 64 + nt * 16 + colo];
            // this D tile = (16 px) x (16 channels of tap t): collision-free consume
#pragma unroll
            for (int r = 0; r < 8; ++r) {
                const int px = pb + r + mhi;
                const float xv = (float)ldsX[(dh * ROWPX + px + dw) * SPX + nt * 16 + colo];
                res[nt][r] += (kv[r] + bv) * xv;
            }
        }
    }

    // ---- store res channels-last bf16 (lanes = contiguous channels) ----
#pragma unroll
    for (int nt = 0; nt < 4; ++nt) {
#pragma unroll
        for (int r = 0; r < 8; ++r) {
            const int px = pb + r + mhi;
            r2[((n * HH + h) * WW + px) * CN + nt * 16 + colo] = (bf16)res[nt][r];
        }
    }
}

// =====================================================================
// k3: conv3x3(res)+b3, relu, channels-last bf16 in -> NCHW f32 out
// (LDS transpose so the NCHW stores are row-contiguous).
// =====================================================================
__global__ __launch_bounds__(256, 2) void MDU_k3_conv(
    const bf16* __restrict__ r2, const unsigned* __restrict__ w3p,
    const float* __restrict__ b3, float* __restrict__ out)
{
    extern __shared__ char smem[];
    bf16*  ldsX = (bf16*)smem;                            // [3*130][SPX]
    float* ldsO = (float*)(smem + 3 * ROWPX * SPX * 2);   // [64][132]

    const int n = blockIdx.x >> 7, h = blockIdx.x & 127;
    const int tid = threadIdx.x;

    // ---- stage channels-last bf16 rows (coalesced: c innermost) ----
    for (int idx = tid; idx < 3 * ROWPX * CN; idx += 256) {
        const int c  = idx & 63;
        const int r  = idx >> 6;
        const int px = r % ROWPX, dh = r / ROWPX;
        const int gh = h + dh - 1, gw = px - 1;
        bf16 vv = (bf16)0.f;
        if (((unsigned)gh < 128u) && ((unsigned)gw < 128u))
            vv = r2[((n * HH + gh) * WW + gw) * CN + c];
        ldsX[(dh * ROWPX + px) * SPX + c] = vv;
    }
    __syncthreads();

    const int lane = tid & 31;
    const int pb   = (tid >> 5) << 4;
    const int mA   = lane & 15;
    const int klb  = (lane >> 4) << 3;
    const int colo = lane & 15;
    const int mhi  = (lane >> 4) << 3;

    v8f acc[4];
#pragma unroll
    for (int i = 0; i < 4; ++i) acc[i] = zero8();

    for (int t = 0; t < 9; ++t) {
        const int dh = t / 3, dw = t % 3;
#pragma unroll
        for (int kc = 0; kc < 2; ++kc) {
            const bf16* ap = ldsX + (dh * ROWPX + pb + mA + dw) * SPX + kc * 32 + klb;
            Frag A;
            A.q[0] = *(const u32x4*)(ap);
            A.q[1] = *(const u32x4*)(ap + 16);
#pragma unroll
            for (int nt = 0; nt < 4; ++nt) {
                const unsigned* bp = w3p + (((t * 2 + kc) * 4 + nt) << 8) + (lane << 3);
                Frag B;
                B.q[0] = *(const u32x4*)(bp);
                B.q[1] = *(const u32x4*)(bp + 4);
                acc[nt] = wmma_bf16(A, B, acc[nt]);
            }
        }
    }

    // ---- bias + relu -> LDS transpose ([c][132] f32, bank-spread) ----
#pragma unroll
    for (int nt = 0; nt < 4; ++nt) {
        const float bv = b3[nt * 16 + colo];
#pragma unroll
        for (int r = 0; r < 8; ++r) {
            const int px = pb + r + mhi;
            ldsO[(nt * 16 + colo) * 132 + px] = fmaxf(acc[nt][r] + bv, 0.f);
        }
    }
    __syncthreads();

    // ---- NCHW f32 store, 512B contiguous per channel-row ----
    for (int idx = tid; idx < CN * WW; idx += 256) {
        const int c = idx >> 7, px = idx & 127;
        out[((n * CN + c) * HH + h) * WW + px] = ldsO[c * 132 + px];
    }
}

// =====================================================================
extern "C" void kernel_launch(void* const* d_in, const int* in_sizes, int n_in,
                              void* d_out, int out_size, void* d_ws, size_t ws_size,
                              hipStream_t stream) {
    (void)in_sizes; (void)n_in; (void)out_size; (void)ws_size;
    const float* x  = (const float*)d_in[0];
    const float* yy = (const float*)d_in[1];
    const float* w1 = (const float*)d_in[2];
    const float* b1 = (const float*)d_in[3];
    const float* w2 = (const float*)d_in[4];
    const float* b2 = (const float*)d_in[5];
    const float* w3 = (const float*)d_in[6];
    const float* b3 = (const float*)d_in[7];

    char* ws = (char*)d_ws;
    unsigned* w1p = (unsigned*)(ws + WS_W1P);
    unsigned* w3p = (unsigned*)(ws + WS_W3P);
    unsigned* w2p = (unsigned*)(ws + WS_W2P);
    float*    b2p = (float*)   (ws + WS_B2P);
    bf16*     r2  = (bf16*)    (ws + WS_R2);

    MDU_prep_pack<<<217, 256, 0, stream>>>(w1, w2, w3, b2, w1p, w2p, w3p, b2p);

    const int grid = 8 * HH;  // one block per (n, h) row
    const unsigned lds12 = 2 * 3 * ROWPX * SPX * 2 + WW * SPX * 2;      // 130,752 B
    const unsigned lds3  = 3 * ROWPX * SPX * 2 + CN * 132 * 4;          //  89,952 B

    MDU_k12_fused<<<grid, 256, lds12, stream>>>(x, yy, w1p, b1, w2p, b2p, r2);
    MDU_k3_conv  <<<grid, 256, lds3,  stream>>>(r2, w3p, b3, (float*)d_out);
}
